// MyCrossAttention_ori_72739566125131
// MI455X (gfx1250) — matrix operand order repaired and checked by reading, not compiled, not run on hardware
//
#include <hip/hip_runtime.h>

#define B_   4
#define N_   4096
#define NS_  2048
#define C_   512
#define H_   8
#define HD_  64

typedef __attribute__((ext_vector_type(16))) __bf16 v16bf;
typedef __attribute__((ext_vector_type(8)))  float  v8f;
typedef __attribute__((ext_vector_type(4)))  unsigned int v4u;
typedef __attribute__((ext_vector_type(8)))  int v8i;
typedef __attribute__((ext_vector_type(4)))  int v4i;

__device__ __forceinline__ unsigned short f2bf(float f) {
  unsigned int u = __float_as_uint(f);
  u += 0x7FFFu + ((u >> 16) & 1u);   // round-to-nearest-even
  return (unsigned short)(u >> 16);
}

// 16 bf16 halves from two 16-byte-aligned 8-element chunks
__device__ __forceinline__ v16bf load16(const unsigned short* p0, const unsigned short* p1) {
  union { uint4 u[2]; v16bf v; } t;
  t.u[0] = *reinterpret_cast<const uint4*>(p0);
  t.u[1] = *reinterpret_cast<const uint4*>(p1);
  return t.v;
}

// ---- f32 -> bf16 bulk convert (vectorized x4) ----
__global__ void cvt_bf16_kernel(const float* __restrict__ src,
                                unsigned short* __restrict__ dst, int n4) {
  int i = blockIdx.x * blockDim.x + threadIdx.x;
  if (i >= n4) return;
  float4 f = reinterpret_cast<const float4*>(src)[i];
  ushort4 o;
  o.x = f2bf(f.x); o.y = f2bf(f.y); o.z = f2bf(f.z); o.w = f2bf(f.w);
  reinterpret_cast<ushort4*>(dst)[i] = o;
}

// ---- NT GEMM  C[M,N] = A[M,K] * W[N,K]^T   (bf16 in, bf16 out, f32 acc) ----
// block = 4 waves; each wave owns a 16(M) x 64(N) strip; block tile = 64x64.
__global__ __launch_bounds__(128) void gemm_nt_bf16_kernel(
    const unsigned short* __restrict__ A,
    const unsigned short* __restrict__ W,
    unsigned short* __restrict__ Cst,
    int M, int Nn, int K, float outScale)
{
  const int wave = threadIdx.x >> 5;
  const int lane = threadIdx.x & 31;
  const int hw   = lane >> 4;          // lane-half
  const int l16  = lane & 15;
  const int m0 = blockIdx.x * 64 + wave * 16;
  const int n0 = blockIdx.y * 64;

  const unsigned short* arow  = A + (size_t)(m0 + l16) * K + hw * 8;
  const unsigned short* wrow  = W + (size_t)(n0 + l16) * K + hw * 8;
  const size_t nstride = (size_t)16 * K;

  v8f acc0 = {}, acc1 = {}, acc2 = {}, acc3 = {};

  for (int k0 = 0; k0 < K; k0 += 32) {
    v16bf af = load16(arow + k0, arow + k0 + 16);
    __builtin_prefetch(arow + k0 + 64, 0, 1);
    const unsigned short* wb = wrow + k0;
    v16bf b0 = load16(wb + 0 * nstride, wb + 0 * nstride + 16);
    v16bf b1 = load16(wb + 1 * nstride, wb + 1 * nstride + 16);
    v16bf b2 = load16(wb + 2 * nstride, wb + 2 * nstride + 16);
    v16bf b3 = load16(wb + 3 * nstride, wb + 3 * nstride + 16);
    acc0 = __builtin_amdgcn_wmma_f32_16x16x32_bf16(false, af, false, b0, (short)0, acc0, false, false);
    acc1 = __builtin_amdgcn_wmma_f32_16x16x32_bf16(false, af, false, b1, (short)0, acc1, false, false);
    acc2 = __builtin_amdgcn_wmma_f32_16x16x32_bf16(false, af, false, b2, (short)0, acc2, false, false);
    acc3 = __builtin_amdgcn_wmma_f32_16x16x32_bf16(false, af, false, b3, (short)0, acc3, false, false);
  }

  #pragma unroll
  for (int v = 0; v < 8; ++v) {
    int row = m0 + v + 8 * hw;
    size_t base = (size_t)row * Nn + n0 + l16;
    Cst[base +  0] = f2bf(acc0[v] * outScale);
    Cst[base + 16] = f2bf(acc1[v] * outScale);
    Cst[base + 32] = f2bf(acc2[v] * outScale);
    Cst[base + 48] = f2bf(acc3[v] * outScale);
  }
}

// TDM descriptor issue: DMA one 16-row x 64-elem bf16 tile (row stride C_)
// into LDS at ldsAddr, padding +16B after every 128B row (LDS row stride 144B).
__device__ __forceinline__ void tdm_load_ktile(unsigned long long gaddr,
                                               unsigned int ldsAddr) {
  v4u g0;
  g0[0] = 1u;                                   // count=1, user mode, no gather
  g0[1] = ldsAddr;                              // lds_addr [63:32]
  g0[2] = (unsigned int)(gaddr & 0xFFFFFFFFull);            // global_addr low
  g0[3] = (unsigned int)((gaddr >> 32) & 0x1FFFFFFull)      // global_addr high
        | (2u << 30);                           // type = 2 (image)
  v8i g1;
  g1[0] = (int)((1u << 16)        // data_size = 2 bytes
        |       (1u << 20)        // pad_enable
        |       (4u << 22)        // pad_interval: 32 DWORDs (=128B)
        |       (3u << 25));      // pad_amount: 4 DWORDs (=16B)
  g1[1] = (int)((unsigned)HD_ << 16);   // tensor_dim0 = 64 (low 16 bits)
  g1[2] = (int)(16u << 16);             // tensor_dim0 hi=0 | tensor_dim1 = 16
  g1[3] = (int)((unsigned)HD_ << 16);   // tensor_dim1 hi=0 | tile_dim0 = 64
  g1[4] = 16;                           // tile_dim1 = 16, tile_dim2 = 0
  g1[5] = C_;                           // tensor_dim0_stride = 512 (low 32)
  g1[6] = 0;
  g1[7] = 0;
  v4i gz4 = {0, 0, 0, 0};
  v8i gz8 = {0, 0, 0, 0, 0, 0, 0, 0};
  // clang-23 / therock form: (g0, g1, g2, g3, g4, cpol)
  __builtin_amdgcn_tensor_load_to_lds(g0, g1, gz4, gz4, gz8, 0);
}

// ---- streaming attention: S[b,h,n] = softmax_row(qh·kh^T) · mask ----
// 4 waves/block share (b,h); key tiles staged to LDS by the Tensor Data Mover
// (double buffered, TENSORcnt-tracked), consumed via conflict-free ds_load_b128.
#define KT_STRIDE 72   // 64 elems + 8 pad shorts (TDM pad: 144B LDS row stride)

__global__ __launch_bounds__(128) void attn_mask_kernel(
    const unsigned short* __restrict__ qh,   // [B,N,C] bf16, pre-scaled by 1/8
    const unsigned short* __restrict__ kh,   // [B,Ns,C] bf16
    const int* __restrict__ mask,            // [B,Ns]
    float* __restrict__ S)                   // [B,H,N]
{
  __shared__ alignas(16) unsigned short kt[2][16 * KT_STRIDE];

  const int wave = threadIdx.x >> 5;
  const int lane = threadIdx.x & 31;
  const int hw = lane >> 4, l16 = lane & 15;

  int idx = blockIdx.x * 4 + wave;           // B*H*(N/16) = 8192 tiles
  int n0 = (idx & 255) * 16;                 // N/16 = 256 tiles per (b,h)
  int bh = idx >> 8;
  int b = bh >> 3, h = bh & 7;

  const unsigned short* qrow = qh + (size_t)(b * N_ + n0 + l16) * C_ + h * HD_;
  v16bf a0 = load16(qrow + hw * 8,      qrow + hw * 8 + 16);
  v16bf a1 = load16(qrow + 32 + hw * 8, qrow + 32 + hw * 8 + 16);

  // TDM source base for this (b,h): tile t starts at key row t*16
  const unsigned short* ksrc = kh + (size_t)(b * NS_) * C_ + h * HD_;
  const unsigned int ldsBase[2] = {
    (unsigned int)(unsigned long long)(uintptr_t)&kt[0][0],
    (unsigned int)(unsigned long long)(uintptr_t)&kt[1][0]
  };
  const int* mrow = mask + b * NS_ + l16;

  float den[8], num[8];
  #pragma unroll
  for (int v = 0; v < 8; ++v) { den[v] = 0.f; num[v] = 0.f; }

  const int NT = NS_ / 16;                   // 128 key tiles
  if (wave == 0)
    tdm_load_ktile((unsigned long long)(uintptr_t)ksrc, ldsBase[0]);

  for (int t = 0; t < NT; ++t) {
    const int cur = t & 1;
    if (wave == 0)
      __builtin_amdgcn_s_wait_tensorcnt(0);  // tile t landed in LDS
    __syncthreads();                          // ready + prior readers done
    if (wave == 0 && (t + 1) < NT)
      tdm_load_ktile((unsigned long long)(uintptr_t)
                         (ksrc + (size_t)(t + 1) * 16 * C_),
                     ldsBase[1 - cur]);

    const unsigned short* kb = &kt[cur][(size_t)l16 * KT_STRIDE + hw * 8];
    v16bf b0 = load16(kb,      kb + 16);      // K = 0..31 of head slice
    v16bf b1 = load16(kb + 32, kb + 48);      // K = 32..63
    v8f acc = {};
    acc = __builtin_amdgcn_wmma_f32_16x16x32_bf16(false, a0, false, b0, (short)0, acc, false, false);
    acc = __builtin_amdgcn_wmma_f32_16x16x32_bf16(false, a1, false, b1, (short)0, acc, false, false);
    float mv = (float)mrow[t * 16];           // this lane's key-column mask
    #pragma unroll
    for (int v = 0; v < 8; ++v) {
      float e = __expf(acc[v]);
      den[v] += e;
      num[v] += e * mv;
    }
  }

  // reduce across the 16 key-columns (lanes within each half-wave)
  #pragma unroll
  for (int v = 0; v < 8; ++v) {
    #pragma unroll
    for (int off = 8; off >= 1; off >>= 1) {
      den[v] += __shfl_xor(den[v], off, 32);
      num[v] += __shfl_xor(num[v], off, 32);
    }
  }
  if (l16 == 0) {
    #pragma unroll
    for (int v = 0; v < 8; ++v)
      S[(size_t)(b * H_ + h) * N_ + n0 + v + 8 * hw] = num[v] / den[v];
  }
}

// ---- Wr[c,h] = sum_d Wp[c, h*64+d]  (rank-8 collapse of final projection) ----
__global__ void wr_kernel(const float* __restrict__ Wp, float* __restrict__ Wr) {
  int i = blockIdx.x * blockDim.x + threadIdx.x;       // 4096 = C*H
  if (i >= C_ * H_) return;
  int c = i >> 3, h = i & 7;
  const float* p = Wp + (size_t)c * C_ + h * HD_;
  float a = 0.f;
  #pragma unroll
  for (int d = 0; d < HD_; ++d) a += p[d];
  Wr[i] = a;
}

// ---- out[b,n,c] = sum_h S[b,h,n] * Wr[c,h] ----
__global__ __launch_bounds__(256) void out_kernel(
    const float* __restrict__ S, const float* __restrict__ Wr,
    float* __restrict__ out)
{
  int row = blockIdx.x;          // b*N + n
  int b = row >> 12;             // N = 4096
  int n = row & 4095;
  float s[8];
  #pragma unroll
  for (int h = 0; h < 8; ++h) s[h] = S[(size_t)(b * H_ + h) * N_ + n];
  for (int c = threadIdx.x; c < C_; c += 256) {
    const float* wr = Wr + c * 8;
    float a = 0.f;
    #pragma unroll
    for (int h = 0; h < 8; ++h) a += s[h] * wr[h];
    out[(size_t)row * C_ + c] = a;
  }
}

extern "C" void kernel_launch(void* const* d_in, const int* in_sizes, int n_in,
                              void* d_out, int out_size, void* d_ws, size_t ws_size,
                              hipStream_t stream) {
  (void)in_sizes; (void)n_in; (void)out_size; (void)ws_size;
  const float* q    = (const float*)d_in[0];
  const float* k    = (const float*)d_in[1];
  const int*   supp = (const int*)  d_in[3];
  const float* Wq   = (const float*)d_in[4];
  const float* Wk   = (const float*)d_in[5];
  const float* Wp   = (const float*)d_in[6];
  float* out = (float*)d_out;

  char* ws = (char*)d_ws;
  size_t off = 0;
  auto alloc = [&](size_t bytes) {
    void* p = ws + off;
    off += (bytes + 255) & ~(size_t)255;
    return p;
  };
  unsigned short* Wq_bf = (unsigned short*)alloc((size_t)C_ * C_ * 2);
  unsigned short* Wk_bf = (unsigned short*)alloc((size_t)C_ * C_ * 2);
  unsigned short* q_bf  = (unsigned short*)alloc((size_t)B_ * N_  * C_ * 2);
  unsigned short* k_bf  = (unsigned short*)alloc((size_t)B_ * NS_ * C_ * 2);
  unsigned short* qh_bf = (unsigned short*)alloc((size_t)B_ * N_  * C_ * 2);
  unsigned short* kh_bf = (unsigned short*)alloc((size_t)B_ * NS_ * C_ * 2);
  float*          Sbuf  = (float*)alloc((size_t)B_ * H_ * N_ * 4);
  float*          Wr    = (float*)alloc((size_t)C_ * H_ * 4);

  // 1) down-convert operands to bf16
  {
    int n4q = B_ * N_  * C_ / 4;   // 2097152
    int n4k = B_ * NS_ * C_ / 4;   // 1048576
    int n4w = C_ * C_ / 4;         // 65536
    cvt_bf16_kernel<<<n4q / 256, 256, 0, stream>>>(q,  q_bf,  n4q);
    cvt_bf16_kernel<<<n4k / 256, 256, 0, stream>>>(k,  k_bf,  n4k);
    cvt_bf16_kernel<<<n4w / 256, 256, 0, stream>>>(Wq, Wq_bf, n4w);
    cvt_bf16_kernel<<<n4w / 256, 256, 0, stream>>>(Wk, Wk_bf, n4w);
  }

  // 2) projections (WMMA bf16); attention scale 1/sqrt(64)=0.125 folded into qh
  gemm_nt_bf16_kernel<<<dim3((B_ * N_)  / 64, C_ / 64), 128, 0, stream>>>(
      q_bf, Wq_bf, qh_bf, B_ * N_,  C_, C_, 0.125f);
  gemm_nt_bf16_kernel<<<dim3((B_ * NS_) / 64, C_ / 64), 128, 0, stream>>>(
      k_bf, Wk_bf, kh_bf, B_ * NS_, C_, C_, 1.0f);

  // 3) streaming masked softmax (TDM-staged keys, WMMA scores, flash sums)
  attn_mask_kernel<<<(B_ * H_ * (N_ / 16)) / 4, 128, 0, stream>>>(
      qh_bf, kh_bf, supp, Sbuf);

  // 4) rank-8 output projection
  wr_kernel<<<(C_ * H_) / 256, 256, 0, stream>>>(Wp, Wr);
  out_kernel<<<B_ * N_, 256, 0, stream>>>(Sbuf, Wr, out);
}
